// Time_bra_43138651521513
// MI455X (gfx1250) — matrix operand order repaired and checked by reading, not compiled, not run on hardware
//
#include <hip/hip_runtime.h>

#define NB   4
#define CD   512
#define TD   1024
#define TDP  (TD + 2)     // padded t rows for k=3 conv
#define NR   512
#define HEADS 8
#define HD   64
#define TOPK 40
#define KROWS 80          // TOPK * rs
#define EPS  1e-5f

typedef __attribute__((ext_vector_type(16))) __bf16 v16bf;
typedef __attribute__((ext_vector_type(8)))  float  v8f;

union ABu { v16bf v; unsigned int u[8]; uint4 q[2]; };

__device__ __forceinline__ unsigned short f2bf(float f) {
    unsigned int u = __float_as_uint(f);
    u += 0x7FFFu + ((u >> 16) & 1u);
    return (unsigned short)(u >> 16);
}

__device__ __forceinline__ void load_fragA(ABu& a, const unsigned short* ap,
                                           int c0, int half) {
    a.q[0] = *(const uint4*)(ap + c0 + half * 8);        // K = kb..kb+7
    a.q[1] = *(const uint4*)(ap + c0 + 16 + half * 8);   // K = 16+kb..16+kb+7
}
__device__ __forceinline__ void load_fragB(ABu& b, const unsigned short* bp,
                                           int c0, int half) {
    b.q[0] = *(const uint4*)(bp + c0 + half * 8);        // K = kb..kb+7
    b.q[1] = *(const uint4*)(bp + c0 + 16 + half * 8);   // K = 16+kb..16+kb+7
}

// ---------------- prep kernels ----------------

// zero the two padding rows (t = 0 and t = TD+1) of padded xT
__global__ __launch_bounds__(256) void k_zero_pad(unsigned short* __restrict__ xT) {
    int tid = blockIdx.x * 256 + threadIdx.x;
    if (tid >= NB * 2 * CD) return;
    int c = tid % CD;
    int e = (tid / CD) % 2;
    int n = tid / (2 * CD);
    int row = e ? (TDP - 1) : 0;
    xT[((size_t)n * TDP + row) * CD + c] = 0;
}

// x (N,C,T) f32 -> xT (N, TD+2, C) bf16, interior rows 1..TD
__global__ __launch_bounds__(256) void k_x_transpose(const float* __restrict__ x,
                                                     unsigned short* __restrict__ xT) {
    int tid = blockIdx.x * 256 + threadIdx.x;
    if (tid >= NB * CD * TD) return;
    int t = tid % TD;
    int c = (tid / TD) % CD;
    int n = tid / (TD * CD);
    xT[((size_t)n * TDP + t + 1) * CD + c] = f2bf(x[tid]);
}

// W (O,I,3) f32 -> Wt (O,3,I) bf16  (K = ci contiguous)
__global__ __launch_bounds__(256) void k_prep_w3(const float* __restrict__ W,
                                                 unsigned short* __restrict__ Wt) {
    int tid = blockIdx.x * 256 + threadIdx.x;
    if (tid >= CD * CD * 3) return;
    int d  = tid % 3;
    int ci = (tid / 3) % CD;
    int o  = tid / (3 * CD);
    Wt[((size_t)o * 3 + d) * CD + ci] = f2bf(W[tid]);
}

// W (O,I,1) f32 -> bf16 (O,I)
__global__ __launch_bounds__(256) void k_prep_w1(const float* __restrict__ W,
                                                 unsigned short* __restrict__ Wt) {
    int tid = blockIdx.x * 256 + threadIdx.x;
    if (tid >= CD * CD) return;
    Wt[tid] = f2bf(W[tid]);
}

// ---------------- conv GEMM (q/k/v and final 1x1 out) ----------------
// One wave computes a 32x32 output tile (2x2 WMMA register blocking).
// B is (N, rowsPerN, C) bf16; for ntaps==3 it is the padded xT and taps index t+d.
// mode 0: out -> (N,T,C) f32 with optional BN;  mode 1: out -> (N,C,T) f32 + bias.
__global__ __launch_bounds__(256) void k_conv_gemm(
    const unsigned short* __restrict__ Wt, int ntaps,
    const unsigned short* __restrict__ Bsrc, int rowsPerN,
    const float* __restrict__ bg, const float* __restrict__ bb,
    const float* __restrict__ bm, const float* __restrict__ bvv,
    float* __restrict__ outT,
    float* __restrict__ outNCT, const float* __restrict__ obias,
    int mode) {
    int wave = threadIdx.x >> 5;
    int lane = threadIdx.x & 31;
    int tile = blockIdx.x * 8 + wave;          // per n: (C/32)*(T/32) = 512 tiles
    int n    = blockIdx.y;
    int tm2  = tile >> 5;                       // 0..15 (32 c_out rows)
    int tt2  = tile & 31;                       // 0..31 (32 t cols)
    int half = lane >> 4;
    int mr   = lane & 15;

    v8f acc[2][2] = {{{}, {}}, {{}, {}}};

    for (int d = 0; d < ntaps; ++d) {
        const unsigned short* ap0 =
            Wt + ((size_t)(tm2 * 32 + mr) * ntaps + d) * CD;
        const unsigned short* ap1 =
            Wt + ((size_t)(tm2 * 32 + 16 + mr) * ntaps + d) * CD;
        const unsigned short* bp0 =
            Bsrc + ((size_t)n * rowsPerN + (tt2 * 32 + mr + d)) * CD;
        const unsigned short* bp1 =
            Bsrc + ((size_t)n * rowsPerN + (tt2 * 32 + 16 + mr + d)) * CD;
        for (int c0 = 0; c0 < CD; c0 += 32) {
            ABu a0, a1, b0, b1;
            load_fragA(a0, ap0, c0, half);
            load_fragA(a1, ap1, c0, half);
            load_fragB(b0, bp0, c0, half);
            load_fragB(b1, bp1, c0, half);
            acc[0][0] = __builtin_amdgcn_wmma_f32_16x16x32_bf16(
                false, a0.v, false, b0.v, (short)0, acc[0][0], false, false);
            acc[0][1] = __builtin_amdgcn_wmma_f32_16x16x32_bf16(
                false, a0.v, false, b1.v, (short)0, acc[0][1], false, false);
            acc[1][0] = __builtin_amdgcn_wmma_f32_16x16x32_bf16(
                false, a1.v, false, b0.v, (short)0, acc[1][0], false, false);
            acc[1][1] = __builtin_amdgcn_wmma_f32_16x16x32_bf16(
                false, a1.v, false, b1.v, (short)0, acc[1][1], false, false);
        }
    }

#pragma unroll
    for (int mi = 0; mi < 2; ++mi) {
#pragma unroll
        for (int ni = 0; ni < 2; ++ni) {
            int tcol = tt2 * 32 + ni * 16 + mr;
#pragma unroll
            for (int r = 0; r < 8; ++r) {
                int c = tm2 * 32 + mi * 16 + r + half * 8;
                float val = acc[mi][ni][r];
                if (mode == 0) {
                    if (bg) {
                        float s = bg[c] * rsqrtf(bvv[c] + EPS);
                        val = (val - bm[c]) * s + bb[c];
                    }
                    outT[((size_t)n * TD + tcol) * CD + c] = val;
                } else {
                    outNCT[((size_t)n * CD + c) * TD + tcol] = val + obias[c];
                }
            }
        }
    }
}

// ---------------- region pooling ----------------
__global__ __launch_bounds__(256) void k_pool(const float* __restrict__ srcT,
                                              unsigned short* __restrict__ dstRT) {
    int tid = blockIdx.x * 256 + threadIdx.x;
    if (tid >= NB * NR * CD) return;
    int c = tid % CD;
    int r = (tid / CD) % NR;
    int n = tid / (CD * NR);
    float a = srcT[((size_t)n * TD + 2 * r) * CD + c];
    float b = srcT[((size_t)n * TD + 2 * r + 1) * CD + c];
    dstRT[tid] = f2bf(0.5f * (a + b));
}

// ---------------- region affinity GEMM (2x2 blocked) ----------------
__global__ __launch_bounds__(256) void k_ar_gemm(const unsigned short* __restrict__ qrT,
                                                 const unsigned short* __restrict__ krT,
                                                 float* __restrict__ ar) {
    int wave = threadIdx.x >> 5;
    int lane = threadIdx.x & 31;
    int tile = blockIdx.x * 8 + wave;     // per n: (NR/32)^2 = 256 tiles
    int n    = blockIdx.y;
    int tr2  = tile >> 4;
    int ts2  = tile & 15;
    int half = lane >> 4;
    int mr   = lane & 15;

    const unsigned short* ap0 = qrT + ((size_t)(n * NR + tr2 * 32 + mr)) * CD;
    const unsigned short* ap1 = ap0 + (size_t)16 * CD;
    const unsigned short* bp0 = krT + ((size_t)(n * NR + ts2 * 32 + mr)) * CD;
    const unsigned short* bp1 = bp0 + (size_t)16 * CD;

    v8f acc[2][2] = {{{}, {}}, {{}, {}}};
    for (int c0 = 0; c0 < CD; c0 += 32) {
        ABu a0, a1, b0, b1;
        load_fragA(a0, ap0, c0, half);
        load_fragA(a1, ap1, c0, half);
        load_fragB(b0, bp0, c0, half);
        load_fragB(b1, bp1, c0, half);
        acc[0][0] = __builtin_amdgcn_wmma_f32_16x16x32_bf16(
            false, a0.v, false, b0.v, (short)0, acc[0][0], false, false);
        acc[0][1] = __builtin_amdgcn_wmma_f32_16x16x32_bf16(
            false, a0.v, false, b1.v, (short)0, acc[0][1], false, false);
        acc[1][0] = __builtin_amdgcn_wmma_f32_16x16x32_bf16(
            false, a1.v, false, b0.v, (short)0, acc[1][0], false, false);
        acc[1][1] = __builtin_amdgcn_wmma_f32_16x16x32_bf16(
            false, a1.v, false, b1.v, (short)0, acc[1][1], false, false);
    }
#pragma unroll
    for (int mi = 0; mi < 2; ++mi) {
#pragma unroll
        for (int ni = 0; ni < 2; ++ni) {
            int scol = ts2 * 32 + ni * 16 + mr;
#pragma unroll
            for (int r = 0; r < 8; ++r) {
                int rrow = tr2 * 32 + mi * 16 + r + half * 8;
                ar[((size_t)(n * NR + rrow)) * NR + scol] = acc[mi][ni][r];
            }
        }
    }
}

// ---------------- top-k (one wave per row) ----------------
__global__ __launch_bounds__(256) void k_topk(const float* __restrict__ ar,
                                              int* __restrict__ idxOut) {
    int wave = threadIdx.x >> 5;
    int lane = threadIdx.x & 31;
    int row  = blockIdx.x * 8 + wave;      // 0..2047
    float vals[16];
#pragma unroll
    for (int i = 0; i < 16; ++i)
        vals[i] = ar[(size_t)row * NR + i * 32 + lane];
    for (int it = 0; it < TOPK; ++it) {
        float best = -3.4e38f;
        int bi = 0x7fffffff;
#pragma unroll
        for (int i = 0; i < 16; ++i) {
            float v = vals[i];
            int gid = i * 32 + lane;
            if (v > best) { best = v; bi = gid; }
        }
#pragma unroll
        for (int off = 16; off > 0; off >>= 1) {
            float ov = __shfl_xor(best, off, 32);
            int   oi = __shfl_xor(bi, off, 32);
            if (ov > best || (ov == best && oi < bi)) { best = ov; bi = oi; }
        }
        if ((bi & 31) == lane) vals[bi >> 5] = -3.4e38f;
        if (lane == 0) idxOut[(size_t)row * TOPK + it] = bi;
    }
}

// ---------------- gathered attention (one wave per (n,h,region)) ----------------
__global__ __launch_bounds__(256) void k_attn(const float* __restrict__ qT,
                                              const float* __restrict__ kT,
                                              const float* __restrict__ vT,
                                              const int* __restrict__ idx,
                                              float* __restrict__ oT) {
    __shared__ float Qs[8 * 128];
    __shared__ float Ss[8 * 160];
    __shared__ int   Tt[8 * KROWS];
    int wave = threadIdx.x >> 5;
    int lane = threadIdx.x & 31;
    int gid  = blockIdx.x * 8 + wave;     // 16384 total
    int r = gid & (NR - 1);
    int h = (gid >> 9) & 7;
    int n = gid >> 12;
    const float SCALE = rsqrtf((float)CD);

    float* Qw = &Qs[wave * 128];
    float* Sw = &Ss[wave * 160];
    int*   Tw = &Tt[wave * KROWS];

    for (int e = lane; e < 128; e += 32) {
        int j = e >> 6, dd = e & 63;
        Qw[e] = qT[((size_t)n * TD + 2 * r + j) * CD + h * HD + dd];
    }
    for (int kk = lane; kk < KROWS; kk += 32) {
        int reg = idx[((size_t)(n * NR + r)) * TOPK + (kk >> 1)];
        Tw[kk] = 2 * reg + (kk & 1);
    }
    __syncthreads();

    for (int kk = lane; kk < KROWS; kk += 32) {
        int t = Tw[kk];
        const float* krow = kT + ((size_t)n * TD + t) * CD + h * HD;
        float s0 = 0.f, s1 = 0.f;
        for (int dd = 0; dd < HD; ++dd) {
            float kv = krow[dd];
            s0 += kv * Qw[dd];
            s1 += kv * Qw[64 + dd];
        }
        Sw[kk]      = s0 * SCALE;
        Sw[80 + kk] = s1 * SCALE;
    }
    __syncthreads();

    if (lane < 2) {
        float* rowp = &Sw[lane * 80];
        float mx = -3.4e38f;
        for (int kk = 0; kk < KROWS; ++kk) mx = fmaxf(mx, rowp[kk]);
        float sum = 0.f;
        for (int kk = 0; kk < KROWS; ++kk) {
            float e = __expf(rowp[kk] - mx);
            rowp[kk] = e; sum += e;
        }
        float inv = 1.0f / sum;
        for (int kk = 0; kk < KROWS; ++kk) rowp[kk] *= inv;
    }
    __syncthreads();

    int dd0 = lane, dd1 = lane + 32;
    float o00 = 0.f, o01 = 0.f, o10 = 0.f, o11 = 0.f;
    for (int kk = 0; kk < KROWS; ++kk) {
        int t = Tw[kk];
        const float* vrow = vT + ((size_t)n * TD + t) * CD + h * HD;
        float va = vrow[dd0], vb = vrow[dd1];
        float p0 = Sw[kk], p1 = Sw[80 + kk];
        o00 += p0 * va; o01 += p0 * vb;
        o10 += p1 * va; o11 += p1 * vb;
    }
    float* orow0 = oT + ((size_t)n * TD + 2 * r) * CD + h * HD;
    float* orow1 = orow0 + CD;
    orow0[dd0] = o00; orow0[dd1] = o01;
    orow1[dd0] = o10; orow1[dd1] = o11;
}

// ---------------- lepe + add, emit bf16 (N,T,C) ----------------
__global__ __launch_bounds__(256) void k_lepe_add(const float* __restrict__ oT,
                                                  const float* __restrict__ vT,
                                                  const float* __restrict__ lw,
                                                  const float* __restrict__ lb,
                                                  unsigned short* __restrict__ oplus) {
    int tid = blockIdx.x * 256 + threadIdx.x;
    if (tid >= NB * TD * CD) return;
    int c = tid & (CD - 1);
    int t = (tid >> 9) & (TD - 1);
    int n = tid >> 19;
    float acc = lb[c];
#pragma unroll
    for (int d = 0; d < 3; ++d) {
        int tt = t + d - 1;
        if (tt >= 0 && tt < TD)
            acc += lw[c * 3 + d] * vT[((size_t)n * TD + tt) * CD + c];
    }
    oplus[tid] = f2bf(oT[tid] + acc);
}

// ---------------- launch ----------------
extern "C" void kernel_launch(void* const* d_in, const int* in_sizes, int n_in,
                              void* d_out, int out_size, void* d_ws, size_t ws_size,
                              hipStream_t stream) {
    (void)in_sizes; (void)n_in; (void)out_size; (void)ws_size;
    const float* x      = (const float*)d_in[0];
    const float* Wq     = (const float*)d_in[1];
    const float* bnq_g  = (const float*)d_in[2];
    const float* bnq_b  = (const float*)d_in[3];
    const float* bnq_m  = (const float*)d_in[4];
    const float* bnq_v  = (const float*)d_in[5];
    const float* Wk     = (const float*)d_in[6];
    const float* bnk_g  = (const float*)d_in[7];
    const float* bnk_b  = (const float*)d_in[8];
    const float* bnk_m  = (const float*)d_in[9];
    const float* bnk_v  = (const float*)d_in[10];
    const float* Wv     = (const float*)d_in[11];
    const float* lepe_w = (const float*)d_in[12];
    const float* lepe_b = (const float*)d_in[13];
    const float* out_w  = (const float*)d_in[14];
    const float* out_b  = (const float*)d_in[15];
    float* out = (float*)d_out;

    char* ws = (char*)d_ws;
    unsigned short* xT   = (unsigned short*)(ws + 0);          // 4*(TD+2)*CD*2 = 4,202,496
    unsigned short* wqT  = (unsigned short*)(ws + 4210688);
    unsigned short* wkT  = (unsigned short*)(ws + 5783552);
    unsigned short* wv   = (unsigned short*)(ws + 7356416);
    unsigned short* ow   = (unsigned short*)(ws + 7880704);
    float* qT            = (float*)(ws + 8404992);             // 8 MiB
    float* kT            = (float*)(ws + 16793600);            // 8 MiB
    float* vT            = (float*)(ws + 25182208);            // 8 MiB
    unsigned short* qrT  = (unsigned short*)(ws + 33570816);   // 2 MiB
    unsigned short* krT  = (unsigned short*)(ws + 35667968);   // 2 MiB
    float* ar            = (float*)(ws + 37765120);            // 4 MiB
    int*   idx           = (int*)(ws + 41959424);              // 320 KiB
    float* oT            = (float*)(ws + 33570816);            // aliases qrT/krT/ar (dead by then)
    unsigned short* oplus= (unsigned short*)(ws + 42287104);   // 4 MiB

    // prep
    k_zero_pad<<<(NB * 2 * CD + 255) / 256, 256, 0, stream>>>(xT);
    k_x_transpose<<<(NB * CD * TD + 255) / 256, 256, 0, stream>>>(x, xT);
    k_prep_w3<<<(CD * CD * 3 + 255) / 256, 256, 0, stream>>>(Wq, wqT);
    k_prep_w3<<<(CD * CD * 3 + 255) / 256, 256, 0, stream>>>(Wk, wkT);
    k_prep_w1<<<(CD * CD + 255) / 256, 256, 0, stream>>>(Wv, wv);
    k_prep_w1<<<(CD * CD + 255) / 256, 256, 0, stream>>>(out_w, ow);

    // q/k/v convs via WMMA (2x2 blocked, 512 tiles/n, 8 waves/block)
    dim3 gconv(64, NB);
    k_conv_gemm<<<gconv, 256, 0, stream>>>(wqT, 3, xT, TDP, bnq_g, bnq_b, bnq_m, bnq_v,
                                           qT, nullptr, nullptr, 0);
    k_conv_gemm<<<gconv, 256, 0, stream>>>(wkT, 3, xT, TDP, bnk_g, bnk_b, bnk_m, bnk_v,
                                           kT, nullptr, nullptr, 0);
    // v: 1x1 conv; use padded xT rows 1..TD by offsetting base one row
    k_conv_gemm<<<gconv, 256, 0, stream>>>(wv, 1, xT + (size_t)CD, TDP,
                                           nullptr, nullptr, nullptr, nullptr,
                                           vT, nullptr, nullptr, 0);

    // region pooling + affinity + top-k
    k_pool<<<(NB * NR * CD + 255) / 256, 256, 0, stream>>>(qT, qrT);
    k_pool<<<(NB * NR * CD + 255) / 256, 256, 0, stream>>>(kT, krT);
    dim3 gar(32, NB);
    k_ar_gemm<<<gar, 256, 0, stream>>>(qrT, krT, ar);
    k_topk<<<256, 256, 0, stream>>>(ar, idx);

    // gathered attention (VALU; tiny FLOP share, irregular gather)
    k_attn<<<2048, 256, 0, stream>>>(qT, kT, vT, idx, oT);

    // lepe + final 1x1 conv via WMMA
    k_lepe_add<<<(NB * TD * CD + 255) / 256, 256, 0, stream>>>(oT, vT, lepe_w, lepe_b, oplus);
    k_conv_gemm<<<gconv, 256, 0, stream>>>(ow, 1, oplus, TD,
                                           nullptr, nullptr, nullptr, nullptr,
                                           nullptr, out, out_b, 1);
}
